// RelativeAttention_70849780515256
// MI455X (gfx1250) — matrix operand order repaired and checked by reading, not compiled, not run on hardware
//
#include <hip/hip_runtime.h>
#include <hip/hip_bf16.h>
#include <math.h>

// ---------------------------------------------------------------------------
// RelativeAttention on MI455X (gfx1250, wave32, WMMA f16 16x16x32 -> f32)
// B=256, S=64, D=1024, H=16, DH=64, BOARD=8, NREL=15
// ---------------------------------------------------------------------------

typedef _Float16 half_t;
typedef __attribute__((ext_vector_type(16))) _Float16 v16h;
typedef __attribute__((ext_vector_type(8)))  float    v8f;

#define NB   256
#define NS   64
#define ND   1024
#define NH   16
#define NDH  64
#define NTOK (NB * NS)           // 16384

// ---------------- WMMA helpers (ISA 7.12.2 layouts) ------------------------

__device__ __forceinline__ v8f wmma_f16(v16h a, v16h b, v8f c) {
  return __builtin_amdgcn_wmma_f32_16x16x32_f16(false, a, false, b, (short)0, c,
                                                false, false);
}

// A-matrix fragment 16x32 f16.  lane: m = lane%16; K runs [b8,b8+8) and
// [b8+16,b8+24), b8 = (lane<16 ? 0 : 8).  Row-major source, 16B-aligned rows.
__device__ __forceinline__ v16h loadA(const half_t* base, int pitch, int row,
                                      int col, int lane) {
  int m  = lane & 15;
  int b8 = (lane < 16) ? 0 : 8;
  const half_t* p = base + (size_t)(row + m) * pitch + col + b8;
  union { v16h v; uint4 q[2]; } f;
  f.q[0] = *(const uint4*)(p);
  f.q[1] = *(const uint4*)(p + 16);
  return f.v;
}

// B-matrix fragment 32x16 f16, sourced from Bt[n][k] (row = output col n).
// lane: n = lane%16; K runs [b8,b8+8) and [b8+16,b8+24), b8 = (lane<16?0:8).
__device__ __forceinline__ v16h loadB(const half_t* base, int pitch, int nbase,
                                      int kbase, int lane) {
  int n  = lane & 15;
  int k0 = kbase + ((lane < 16) ? 0 : 8);
  const half_t* p = base + (size_t)(nbase + n) * pitch + k0;
  union { v16h v; uint4 q[2]; } f;
  f.q[0] = *(const uint4*)(p);
  f.q[1] = *(const uint4*)(p + 16);
  return f.v;
}

// ---------------- conversion / table kernels -------------------------------

__global__ void cvt_f32_to_f16(const float* __restrict__ src,
                               half_t* __restrict__ dst, int n) {
  for (int i = blockIdx.x * blockDim.x + threadIdx.x; i < n;
       i += gridDim.x * blockDim.x)
    dst[i] = (half_t)src[i];
}

// wcat16: rows 0..1023 wq, 1024..2047 wk, 2048..3071 wv, 3072..4095 wo
__global__ void cvt_weights(const float* __restrict__ wq,
                            const float* __restrict__ wk,
                            const float* __restrict__ wv,
                            const float* __restrict__ wo,
                            half_t* __restrict__ wcat) {
  const int n = 4096 * 1024;
  for (int i = blockIdx.x * blockDim.x + threadIdx.x; i < n;
       i += gridDim.x * blockDim.x) {
    int row = i >> 10, col = i & 1023;
    const float* s = (row < 1024) ? wq : (row < 2048) ? wk
                     : (row < 3072) ? wv : wo;
    wcat[i] = (half_t)s[((row & 1023) << 10) + col];
  }
}

// tabs: [ekh | ekw | eqh | eqw] each 16x64 (row 15 zero-padded), f16
// evT:  64x32 f16, evT[d][r] = ev_h[r][d] (r<15), ev_w[r-16][d] (16<=r<31)
// R2:   64x64 f32, R2[i][j] = sum_d rel_q[i,j,d]*rel_k[i,j,d]
__global__ void build_tabs(const float* __restrict__ eqh,
                           const float* __restrict__ eqw,
                           const float* __restrict__ ekh,
                           const float* __restrict__ ekw,
                           const float* __restrict__ evh,
                           const float* __restrict__ evw,
                           half_t* __restrict__ tabs,
                           half_t* __restrict__ evT,
                           float* __restrict__ R2) {
  int idx = blockIdx.x * blockDim.x + threadIdx.x;  // 0..10239
  if (idx < 4096) {
    int tbl = idx >> 10, r = (idx >> 6) & 15, d = idx & 63;
    const float* s = (tbl == 0) ? ekh : (tbl == 1) ? ekw
                     : (tbl == 2) ? eqh : eqw;
    tabs[idx] = (half_t)((r < 15) ? s[r * 64 + d] : 0.0f);
  } else if (idx < 6144) {
    int k = idx - 4096, d = k >> 5, r = k & 31;
    float v = 0.0f;
    if (r < 15)                 v = evh[r * 64 + d];
    else if (r >= 16 && r < 31) v = evw[(r - 16) * 64 + d];
    evT[k] = (half_t)v;
  } else if (idx < 10240) {
    int k = idx - 6144, i = k >> 6, j = k & 63;
    int rr = (i >> 3) - (j >> 3) + 7;
    int rc = (i & 7)  - (j & 7)  + 7;
    float a = 0.0f;
#pragma unroll 8
    for (int d = 0; d < 64; ++d)
      a += (eqh[rr * 64 + d] + eqw[rc * 64 + d]) *
           (ekh[rr * 64 + d] + ekw[rc * 64 + d]);
    R2[k] = a;
  }
}

// ---------------- QKV projection GEMM --------------------------------------
// Y = x @ Wcat^T  (16384x1024) x (1024x3072); output f16 scattered to
// head-major (B,H,S,DH) q/k/v buffers.  Block = 128 thr (4 waves), tile 64x64.
__global__ __launch_bounds__(128) void gemm_qkv(
    const half_t* __restrict__ x16, const half_t* __restrict__ w16,
    half_t* __restrict__ q16, half_t* __restrict__ k16,
    half_t* __restrict__ v16) {
  __shared__ __align__(16) half_t sA[64 * 40];
  const int t = threadIdx.x, lane = t & 31, w = t >> 5;
  const int hi = lane >> 4, n16 = lane & 15;
  const int rb = blockIdx.x, cb = blockIdx.y;

  v8f acc[4] = {};
  for (int ks = 0; ks < 1024; ks += 32) {
    {  // stage 64x32 A panel (f16, conflict-friendly pitch 40)
      int row = t >> 1, c0 = (t & 1) * 16;
      const uint4* src =
          (const uint4*)(x16 + (size_t)(rb * 64 + row) * 1024 + ks + c0);
      uint4* dst = (uint4*)(sA + row * 40 + c0);
      dst[0] = src[0];
      dst[1] = src[1];
    }
    __syncthreads();
    v16h a = loadA(sA, 40, w * 16, 0, lane);
#pragma unroll
    for (int jb = 0; jb < 4; ++jb) {
      v16h b = loadB(w16, 1024, cb * 64 + jb * 16, ks, lane);
      acc[jb] = wmma_f16(a, b, acc[jb]);
    }
    __syncthreads();
  }
#pragma unroll
  for (int jb = 0; jb < 4; ++jb) {
#pragma unroll
    for (int r = 0; r < 8; ++r) {
      int gcol  = cb * 64 + jb * 16 + n16;           // 0..3071
      int token = rb * 64 + w * 16 + r + 8 * hi;     // 0..16383
      int which = gcol >> 10, wc = gcol & 1023;
      int hh = wc >> 6, dh = wc & 63;
      half_t* dst = (which == 0) ? q16 : (which == 1) ? k16 : v16;
      size_t o = (((size_t)(token >> 6) * 16 + hh) * 64 + (token & 63)) * 64 + dh;
      dst[o] = (half_t)acc[jb][r];
    }
  }
}

// ---------------- fused relative attention ---------------------------------
// One workgroup per (b,h); 128 threads = 4 waves; S=DH=64.
__global__ __launch_bounds__(128) void attn_kernel(
    const half_t* __restrict__ q16, const half_t* __restrict__ k16,
    const half_t* __restrict__ v16, const half_t* __restrict__ tabs,
    const half_t* __restrict__ evT, const float* __restrict__ R2,
    half_t* __restrict__ ctx16) {
  __shared__ __align__(16) unsigned char smem[60416];
  half_t* sQ  = (half_t*)smem;                 // 64x72 f16 (later: probs)
  half_t* sK  = sQ + 64 * 72;                  // 64x72 f16
  half_t* sVt = sK + 64 * 72;                  // 64x72 f16, transposed v
  float*  sS  = (float*)(smem + 27648);        // 64x64 f32 scores
  float*  sQh = (float*)(smem + 44032);        // 64x16 f32
  float*  sQw = sQh + 1024;
  float*  sKh = sQw + 1024;
  float*  sKw = sKh + 1024;
  half_t* sAB = (half_t*)(smem + 44032);       // 64x40 f16 (overlays sQh/sQw)

  const int bh = blockIdx.x;
  const int b = bh >> 4, h = bh & 15;
  const int t = threadIdx.x, lane = t & 31, w = t >> 5;
  const int hi = lane >> 4, n16 = lane & 15;
  const half_t* tEkh = tabs;
  const half_t* tEkw = tabs + 1024;
  const half_t* tEqh = tabs + 2048;
  const half_t* tEqw = tabs + 3072;

  // --- cooperative loads -----------------------------------------------
  {
    int row = t >> 1, c0 = (t & 1) * 32;
    const uint4* sq = (const uint4*)(q16 + (size_t)bh * 4096 + row * 64 + c0);
    const uint4* sk = (const uint4*)(k16 + (size_t)bh * 4096 + row * 64 + c0);
    uint4* dq = (uint4*)(sQ + row * 72 + c0);
    uint4* dk = (uint4*)(sK + row * 72 + c0);
#pragma unroll
    for (int u = 0; u < 4; ++u) { dq[u] = sq[u]; dk[u] = sk[u]; }
  }
  for (int idx = t; idx < 4096; idx += 128) {
    int j = idx >> 6, d = idx & 63;
    sVt[d * 72 + j] = v16[(size_t)bh * 4096 + idx];  // transpose into LDS
  }
  __syncthreads();

  // --- phase A: scores + skew projections (WMMA) -----------------------
  v8f accS[4] = {};
  v8f aQh = {}, aQw = {}, aKh = {}, aKw = {};
#pragma unroll
  for (int ks = 0; ks < 64; ks += 32) {
    v16h aQ = loadA(sQ, 72, w * 16, ks, lane);
    v16h aK = loadA(sK, 72, w * 16, ks, lane);
#pragma unroll
    for (int jb = 0; jb < 4; ++jb)
      accS[jb] = wmma_f16(aQ, loadB(sK, 72, jb * 16, ks, lane), accS[jb]);
    aQh = wmma_f16(aQ, loadB(tEkh, 64, 0, ks, lane), aQh);
    aQw = wmma_f16(aQ, loadB(tEkw, 64, 0, ks, lane), aQw);
    aKh = wmma_f16(aK, loadB(tEqh, 64, 0, ks, lane), aKh);
    aKw = wmma_f16(aK, loadB(tEqw, 64, 0, ks, lane), aKw);
  }
#pragma unroll
  for (int r = 0; r < 8; ++r) {
    int row = w * 16 + r + 8 * hi;
#pragma unroll
    for (int jb = 0; jb < 4; ++jb)
      sS[row * 64 + jb * 16 + n16] = accS[jb][r];
    sQh[row * 16 + n16] = aQh[r];
    sQw[row * 16 + n16] = aQw[r];
    sKh[row * 16 + n16] = aKh[r];
    sKw[row * 16 + n16] = aKw[r];
  }
  __syncthreads();

  // --- phase B: assemble scores, softmax, probs (thread = row) ---------
  float inv = 0.0f;
  if (t < 64) {
    const int i = t, ri = i >> 3, ci = i & 7;
    float mx = -3.0e38f;
#pragma unroll 8
    for (int j = 0; j < 64; ++j) {
      int rr = ri - (j >> 3) + 7, rc = ci - (j & 7) + 7;
      float val = sS[i * 64 + j] + sQh[i * 16 + rr] + sQw[i * 16 + rc] +
                  sKh[j * 16 + rr] + sKw[j * 16 + rc] + R2[i * 64 + j];
      val *= 0.125f;  // 1/sqrt(64)
      sS[i * 64 + j] = val;
      mx = fmaxf(mx, val);
    }
    float sum = 0.0f;
#pragma unroll 8
    for (int j = 0; j < 64; ++j) {
      float p = __expf(sS[i * 64 + j] - mx);
      sS[i * 64 + j] = p;
      sum += p;
    }
    inv = 1.0f / sum;
#pragma unroll 8
    for (int j = 0; j < 64; ++j)
      sQ[i * 72 + j] = (half_t)(sS[i * 64 + j] * inv);  // normalized probs
  }
  __syncthreads();  // everyone done reading sQh..sKw before sAB overlays them

  // --- phase C: attention bins -> packed [Ah|Aw] A-matrix --------------
  if (t < 64) {
    const int i = t, ri = i >> 3, ci = i & 7;
#pragma unroll
    for (int kk = 0; kk < 32; ++kk) sAB[i * 40 + kk] = (half_t)0;
#pragma unroll
    for (int jb = 0; jb < 8; ++jb) {  // row-block sums -> Ah bins
      float s = 0.0f;
#pragma unroll
      for (int u = 0; u < 8; ++u) s += sS[i * 64 + jb * 8 + u];
      sAB[i * 40 + (ri + 7 - jb)] = (half_t)(s * inv);
    }
#pragma unroll
    for (int cj = 0; cj < 8; ++cj) {  // column-phase sums -> Aw bins
      float s = 0.0f;
#pragma unroll
      for (int jb = 0; jb < 8; ++jb) s += sS[i * 64 + jb * 8 + cj];
      sAB[i * 40 + 16 + (ci + 7 - cj)] = (half_t)(s * inv);
    }
  }
  __syncthreads();

  // --- phase D: ctx = probs @ V + [Ah|Aw] @ evT (WMMA) -----------------
  v8f accC[4] = {};
#pragma unroll
  for (int ks = 0; ks < 64; ks += 32) {
    v16h aP = loadA(sQ, 72, w * 16, ks, lane);
#pragma unroll
    for (int db = 0; db < 4; ++db)
      accC[db] = wmma_f16(aP, loadB(sVt, 72, db * 16, ks, lane), accC[db]);
  }
  {
    v16h aR = loadA(sAB, 40, w * 16, 0, lane);
#pragma unroll
    for (int db = 0; db < 4; ++db)
      accC[db] = wmma_f16(aR, loadB(evT, 32, db * 16, 0, lane), accC[db]);
  }
#pragma unroll
  for (int r = 0; r < 8; ++r) {
    int i = w * 16 + r + 8 * hi;
    size_t tok = (size_t)b * 64 + i;
#pragma unroll
    for (int db = 0; db < 4; ++db)
      ctx16[tok * 1024 + h * 64 + db * 16 + n16] = (half_t)accC[db][r];
  }
}

// ---------------- output projection GEMM (+bias, f32 out) ------------------
__global__ __launch_bounds__(128) void gemm_out(
    const half_t* __restrict__ ctx16, const half_t* __restrict__ wo16,
    const float* __restrict__ bo, float* __restrict__ out) {
  __shared__ __align__(16) half_t sA[64 * 40];
  const int t = threadIdx.x, lane = t & 31, w = t >> 5;
  const int hi = lane >> 4, n16 = lane & 15;
  const int rb = blockIdx.x, cb = blockIdx.y;

  v8f acc[4] = {};
  for (int ks = 0; ks < 1024; ks += 32) {
    {
      int row = t >> 1, c0 = (t & 1) * 16;
      const uint4* src =
          (const uint4*)(ctx16 + (size_t)(rb * 64 + row) * 1024 + ks + c0);
      uint4* dst = (uint4*)(sA + row * 40 + c0);
      dst[0] = src[0];
      dst[1] = src[1];
    }
    __syncthreads();
    v16h a = loadA(sA, 40, w * 16, 0, lane);
#pragma unroll
    for (int jb = 0; jb < 4; ++jb) {
      v16h b = loadB(wo16, 1024, cb * 64 + jb * 16, ks, lane);
      acc[jb] = wmma_f16(a, b, acc[jb]);
    }
    __syncthreads();
  }
#pragma unroll
  for (int jb = 0; jb < 4; ++jb) {
#pragma unroll
    for (int r = 0; r < 8; ++r) {
      int gcol  = cb * 64 + jb * 16 + n16;
      int token = rb * 64 + w * 16 + r + 8 * hi;
      out[(size_t)token * 1024 + gcol] = acc[jb][r] + bo[gcol];
    }
  }
}

// ---------------------------------------------------------------------------

extern "C" void kernel_launch(void* const* d_in, const int* in_sizes, int n_in,
                              void* d_out, int out_size, void* d_ws,
                              size_t ws_size, hipStream_t stream) {
  const float* x   = (const float*)d_in[0];
  const float* wq  = (const float*)d_in[1];
  const float* wk  = (const float*)d_in[2];
  const float* wv  = (const float*)d_in[3];
  const float* wo  = (const float*)d_in[4];
  const float* bo  = (const float*)d_in[5];
  const float* eqh = (const float*)d_in[6];
  const float* eqw = (const float*)d_in[7];
  const float* ekh = (const float*)d_in[8];
  const float* ekw = (const float*)d_in[9];
  const float* evh = (const float*)d_in[10];
  const float* evw = (const float*)d_in[11];
  float* out = (float*)d_out;

  unsigned char* ws = (unsigned char*)d_ws;
  const size_t SZ_TOKD = (size_t)NTOK * ND * sizeof(half_t);  // 33.5 MB
  size_t off = 0;
  half_t* x16   = (half_t*)(ws + off); off += SZ_TOKD;
  half_t* wcat  = (half_t*)(ws + off); off += (size_t)4096 * 1024 * sizeof(half_t);
  half_t* q16   = (half_t*)(ws + off); off += SZ_TOKD;
  half_t* k16   = (half_t*)(ws + off); off += SZ_TOKD;
  half_t* v16   = (half_t*)(ws + off); off += SZ_TOKD;
  half_t* ctx16 = (half_t*)(ws + off); off += SZ_TOKD;
  half_t* tabs  = (half_t*)(ws + off); off += 4096 * sizeof(half_t);
  half_t* evT   = (half_t*)(ws + off); off += 2048 * sizeof(half_t);
  float*  R2    = (float*)(ws + off);  off += 4096 * sizeof(float);
  half_t* wo16  = wcat + (size_t)3072 * 1024;

  cvt_f32_to_f16<<<8192, 256, 0, stream>>>(x, x16, NTOK * ND);
  cvt_weights<<<4096, 256, 0, stream>>>(wq, wk, wv, wo, wcat);
  build_tabs<<<40, 256, 0, stream>>>(eqh, eqw, ekh, ekw, evh, evw, tabs, evT, R2);

  gemm_qkv<<<dim3(NTOK / 64, 3072 / 64), 128, 0, stream>>>(x16, wcat, q16, k16, v16);
  attn_kernel<<<NB * NH, 128, 0, stream>>>(q16, k16, v16, tabs, evT, R2, ctx16);
  gemm_out<<<dim3(NTOK / 64, ND / 64), 128, 0, stream>>>(ctx16, wo16, bo, out);
}
